// PyramidKVModel_40707700031611
// MI455X (gfx1250) — matrix operand order repaired and checked by reading, not compiled, not run on hardware
//
#include <hip/hip_runtime.h>
#include <hip/hip_bf16.h>

// ---------------------------------------------------------------------------
// Types / WMMA helpers (CDNA5 gfx1250, wave32)
// ---------------------------------------------------------------------------
typedef __bf16 bf16_t;
typedef __attribute__((ext_vector_type(16))) __bf16 v16bf;
typedef __attribute__((ext_vector_type(8)))  float  v8f;

union BF16Frag { v16bf v; unsigned int u[8]; };

__device__ __forceinline__ v8f v8f_zero() {
  v8f z;
#pragma unroll
  for (int i = 0; i < 8; ++i) z[i] = 0.0f;
  return z;
}

// A-matrix fragment (16x32 bf16, M x K). LDS layout: row-major [m][k], ld even.
__device__ __forceinline__ v16bf load_a_frag(const bf16_t* base, int ld, int row0,
                                             int kbase, int lane) {
  BF16Frag f;
  const int m  = row0 + (lane & 15);
  const int kh = (lane >> 4) << 3;  // 0 or 8
  const unsigned int* p =
      reinterpret_cast<const unsigned int*>(base + m * ld + kbase);
#pragma unroll
  for (int j = 0; j < 4; ++j) f.u[j]     = p[(kh >> 1) + j];      // k = kh + 2j
#pragma unroll
  for (int j = 0; j < 4; ++j) f.u[4 + j] = p[8 + (kh >> 1) + j];  // k = 16+kh+2j
  return f.v;
}

// B-matrix fragment (32x16 bf16, K x N). LDS layout transposed: [n][k], ld even.
__device__ __forceinline__ v16bf load_b_frag(const bf16_t* base, int ld, int col0,
                                             int kbase, int lane) {
  BF16Frag f;
  const int n  = col0 + (lane & 15);
  const int k0 = kbase + ((lane >> 4) << 3);
  const unsigned int* p =
      reinterpret_cast<const unsigned int*>(base + n * ld + k0);
#pragma unroll
  for (int j = 0; j < 4; ++j) f.u[j] = p[j];
#pragma unroll
  for (int j = 0; j < 4; ++j) f.u[4 + j] = p[8 + j];
  return f.v;
}

#define WMMA_BF16(A, B, C) \
  __builtin_amdgcn_wmma_f32_16x16x32_bf16(false, (A), false, (B), (short)0, (C), false, false)

// ---------------------------------------------------------------------------
// gfx1250 async global->LDS (ASYNCcnt-tracked, CDNA5 ISA)
// ---------------------------------------------------------------------------
__device__ __forceinline__ void async_b128(const void* g, const void* l) {
  unsigned loff = (unsigned)(unsigned long long)l;       // low 32 bits = LDS offset
  unsigned long long ga = (unsigned long long)g;
  asm volatile("global_load_async_to_lds_b128 %0, %1, off"
               :: "v"(loff), "v"(ga) : "memory");
}

__device__ __forceinline__ void wait_async_all() {
  asm volatile("s_wait_asynccnt 0x0" ::: "memory");
}

// 2-D tile copy: rows x rowelems bf16, global row stride gstride elems.
__device__ __forceinline__ void async_copy_2d(const bf16_t* g, int gstride, bf16_t* l,
                                              int lstride, int rows, int rowelems, int tid) {
  const int cpr = rowelems >> 3;  // 16B chunks per row
  const int total = rows * cpr;
  for (int c = tid; c < total; c += 256) {
    int r = c / cpr;
    int s = (c - r * cpr) << 3;
    async_b128(g + (size_t)r * gstride + s, l + r * lstride + s);
  }
}

// ---------------------------------------------------------------------------
// Weight convert + transpose: Wt[n*K + k] = bf16(W[k*N + n])
// ---------------------------------------------------------------------------
__global__ void transpose_w_kernel(const float* __restrict__ W, bf16_t* __restrict__ Wt,
                                   int K, int N) {
  int i = blockIdx.x * blockDim.x + threadIdx.x;
  if (i >= K * N) return;
  int k = i % K, n = i / K;
  Wt[i] = (bf16_t)W[(size_t)k * N + n];
}

// ---------------------------------------------------------------------------
// Embedding: x = bf16(tok_emb[ids] + pos_emb)
// ---------------------------------------------------------------------------
__global__ void embed_kernel(const int* __restrict__ ids, const float* __restrict__ tok,
                             const float* __restrict__ pos, bf16_t* __restrict__ x,
                             int S_, int E) {
  int i = blockIdx.x * blockDim.x + threadIdx.x;
  if (i >= 2 * S_ * E) return;
  int d = i % E;
  int t = i / E;
  int s = t % S_;
  x[i] = (bf16_t)(tok[(size_t)ids[t] * E + d] + pos[(size_t)s * E + d]);
}

// ---------------------------------------------------------------------------
// Async double-buffered bf16 WMMA GEMM.
// A: bf16 [M][K]; Wt: bf16 [N][K] (pre-transposed). Tile 128x128, BK=32,
// 256 threads = 8 wave32 waves, each 32x64 (8 WMMA / K-step).
// EPI: 0 = f32 out, 1 = f32 out + resid, 2 = GELU -> bf16 out,
//      3 = bf16 out packed per-head [B][NH][S][64]
// ---------------------------------------------------------------------------
template <int EPI>
__global__ __launch_bounds__(256) void gemm_async_kernel(
    const bf16_t* __restrict__ A, const bf16_t* __restrict__ Wt,
    const float* __restrict__ bias, const float* __restrict__ resid,
    float* __restrict__ Cf, bf16_t* __restrict__ Cb,
    int M, int N, int K, int S_) {
  __shared__ alignas(16) bf16_t As[2][128 * 32];
  __shared__ alignas(16) bf16_t Bs[2][128 * 32];

  const int tid = threadIdx.x;
  const int lane = tid & 31, wave = tid >> 5;
  const int n0 = blockIdx.x * 128, m0 = blockIdx.y * 128;
  const int wm = wave >> 1;  // 0..3 -> 32-row strip
  const int wn = wave & 1;   // 0..1 -> 64-col strip
  const bf16_t* Abase = A + (size_t)m0 * K;
  const bf16_t* Bbase = Wt + (size_t)n0 * K;

  v8f acc[2][4];
#pragma unroll
  for (int im = 0; im < 2; ++im)
#pragma unroll
    for (int jn = 0; jn < 4; ++jn) acc[im][jn] = v8f_zero();

  const int nk = K / 32;
  async_copy_2d(Abase, K, As[0], 32, 128, 32, tid);
  async_copy_2d(Bbase, K, Bs[0], 32, 128, 32, tid);

  for (int kt = 0; kt < nk; ++kt) {
    const int cur = kt & 1;
    wait_async_all();        // this wave's chunks of tile kt have landed
    __syncthreads();         // whole tile visible; prev-buffer readers done
    if (kt + 1 < nk) {       // prefetch next tile into the other buffer
      async_copy_2d(Abase + (kt + 1) * 32, K, As[cur ^ 1], 32, 128, 32, tid);
      async_copy_2d(Bbase + (kt + 1) * 32, K, Bs[cur ^ 1], 32, 128, 32, tid);
    }
    v16bf a0 = load_a_frag(As[cur], 32, wm * 32 + 0, 0, lane);
    v16bf a1 = load_a_frag(As[cur], 32, wm * 32 + 16, 0, lane);
#pragma unroll
    for (int jn = 0; jn < 4; ++jn) {
      v16bf bfr = load_b_frag(Bs[cur], 32, wn * 64 + jn * 16, 0, lane);
      acc[0][jn] = WMMA_BF16(a0, bfr, acc[0][jn]);
      acc[1][jn] = WMMA_BF16(a1, bfr, acc[1][jn]);
    }
  }

  const int rh = (lane >> 4) * 8;
#pragma unroll
  for (int im = 0; im < 2; ++im)
#pragma unroll
    for (int jn = 0; jn < 4; ++jn)
#pragma unroll
      for (int r = 0; r < 8; ++r) {
        int row = m0 + wm * 32 + im * 16 + r + rh;
        int col = n0 + wn * 64 + jn * 16 + (lane & 15);
        float v = acc[im][jn][r] + bias[col];
        if (EPI == 2) v = 0.5f * v * (1.0f + erff(v * 0.70710678118654752f));
        if (EPI == 1) v += resid[(size_t)row * N + col];
        if (EPI == 0 || EPI == 1) Cf[(size_t)row * N + col] = v;
        if (EPI == 2) Cb[(size_t)row * N + col] = (bf16_t)v;
        if (EPI == 3) {
          int b = row / S_, s = row % S_;
          int h = col >> 6, d = col & 63;
          Cb[(((size_t)(b * 16 + h) * S_) + s) * 64 + d] = (bf16_t)v;
        }
      }
}

// ---------------------------------------------------------------------------
// Row LayerNorm (f32 in, bf16 out), one block per row
// ---------------------------------------------------------------------------
__global__ __launch_bounds__(256) void layernorm_kernel(
    const float* __restrict__ x, const float* __restrict__ sc,
    const float* __restrict__ bi, bf16_t* __restrict__ o, int N) {
  __shared__ float red[256];
  int row = blockIdx.x, tid = threadIdx.x;
  const float* xr = x + (size_t)row * N;
  float sum = 0.0f, sq = 0.0f;
  for (int i = tid; i < N; i += 256) { float v = xr[i]; sum += v; sq += v * v; }
  red[tid] = sum; __syncthreads();
  for (int off = 128; off; off >>= 1) { if (tid < off) red[tid] += red[tid + off]; __syncthreads(); }
  float mean = red[0] / N; __syncthreads();
  red[tid] = sq; __syncthreads();
  for (int off = 128; off; off >>= 1) { if (tid < off) red[tid] += red[tid + off]; __syncthreads(); }
  float var = red[0] / N - mean * mean;
  float inv = rsqrtf(var + 1e-5f);
  for (int i = tid; i < N; i += 256)
    o[(size_t)row * N + i] = (bf16_t)((xr[i] - mean) * inv * sc[i] + bi[i]);
}

// ---------------------------------------------------------------------------
// PyramidKV importance: imp[b][k] += sum_q |(QK^T)[q][k]| (monotonic vs. ref)
// ---------------------------------------------------------------------------
__global__ __launch_bounds__(256) void importance_kernel(
    const bf16_t* __restrict__ QH, const bf16_t* __restrict__ KH,
    float* __restrict__ imp, int S_) {
  __shared__ alignas(16) bf16_t Qs[128 * 64];  // [q][d]
  __shared__ alignas(16) bf16_t Kc[32 * 64];   // [key][d]
  int nq = S_ / 128;
  int qt = blockIdx.x % nq, bh = blockIdx.x / nq;
  int bb = bh >> 4;
  const bf16_t* Q  = QH + ((size_t)bh * S_ + qt * 128) * 64;
  const bf16_t* Kp = KH + (size_t)bh * S_ * 64;
  int tid = threadIdx.x, lane = tid & 31, wave = tid >> 5;

  async_copy_2d(Q, 64, Qs, 64, 128, 64, tid);

  for (int kc = 0; kc < S_; kc += 32) {
    __syncthreads();  // all waves done with previous chunk
    async_copy_2d(Kp + (size_t)kc * 64, 64, Kc, 64, 32, 64, tid);
    wait_async_all();
    __syncthreads();
    v16bf a0 = load_a_frag(Qs, 64, wave * 16, 0, lane);
    v16bf a1 = load_a_frag(Qs, 64, wave * 16, 32, lane);
#pragma unroll
    for (int t = 0; t < 2; ++t) {
      v8f s = v8f_zero();
      v16bf b0 = load_b_frag(Kc, 64, t * 16, 0, lane);
      v16bf b1 = load_b_frag(Kc, 64, t * 16, 32, lane);
      s = WMMA_BF16(a0, b0, s);
      s = WMMA_BF16(a1, b1, s);
      float cs = 0.0f;
#pragma unroll
      for (int r = 0; r < 8; ++r) cs += fabsf(s[r]);
      cs += __shfl_xor(cs, 16, 32);  // rows 0-7 + rows 8-15 (same key col)
      if (lane < 16) atomicAdd(&imp[(size_t)bb * S_ + kc + t * 16 + lane], cs);
    }
  }
}

// ---------------------------------------------------------------------------
// Threshold-bisection top-k (replaces 1500-round serial argmax).
// Invariant: count(v > lo) >= keep > count(v > hi). Order-free set is valid
// since attention is permutation-invariant over selected keys.
// ---------------------------------------------------------------------------
__global__ __launch_bounds__(256) void topk_kernel(const float* __restrict__ imp,
                                                   int* __restrict__ idx, int S_, int keep) {
  __shared__ float vals[2048];
  __shared__ float redf[256];
  __shared__ int   redi[256];
  __shared__ int   cnt;
  int b = blockIdx.x, tid = threadIdx.x;
  for (int i = tid; i < S_; i += 256) vals[i] = imp[(size_t)b * S_ + i];
  __syncthreads();

  float mx = -1e30f;
  for (int i = tid; i < S_; i += 256) mx = fmaxf(mx, vals[i]);
  redf[tid] = mx; __syncthreads();
  for (int off = 128; off; off >>= 1) {
    if (tid < off) redf[tid] = fmaxf(redf[tid], redf[tid + off]);
    __syncthreads();
  }
  float hi = redf[0];   // count(v > max) == 0 < keep
  float lo = -1.0f;     // importance >= 0 -> count == S_ >= keep
  __syncthreads();

  for (int it = 0; it < 40; ++it) {
    float mid = 0.5f * (lo + hi);
    int c = 0;
    for (int i = tid; i < S_; i += 256) c += (vals[i] > mid) ? 1 : 0;
    redi[tid] = c; __syncthreads();
    for (int off = 128; off; off >>= 1) {
      if (tid < off) redi[tid] += redi[tid + off];
      __syncthreads();
    }
    int total = redi[0]; __syncthreads();
    if (total >= keep) lo = mid; else hi = mid;
  }

  if (tid == 0) cnt = 0;
  __syncthreads();
  for (int i = tid; i < S_; i += 256)
    if (vals[i] > hi) { int slot = atomicAdd(&cnt, 1); idx[(size_t)b * S_ + slot] = i; }
  __syncthreads();  // phase-1 slots (< keep guaranteed) all assigned first
  for (int i = tid; i < S_; i += 256)
    if (vals[i] <= hi && vals[i] > lo) {
      int slot = atomicAdd(&cnt, 1);
      if (slot < keep) idx[(size_t)b * S_ + slot] = i;
    }
}

// ---------------------------------------------------------------------------
// Gather selected K/V rows into contiguous bf16 buffers
// ---------------------------------------------------------------------------
__global__ void gather_kv_kernel(const bf16_t* __restrict__ KH, const bf16_t* __restrict__ VH,
                                 const int* __restrict__ idx, bf16_t* __restrict__ KG,
                                 bf16_t* __restrict__ VG, int S_, int keep) {
  int i = blockIdx.x * blockDim.x + threadIdx.x;
  if (i >= 2 * 16 * keep * 64) return;
  int d = i & 63;
  int j = (i >> 6) % keep;
  int bh = i / (64 * keep);
  int b = bh >> 4;
  int src = idx[(size_t)b * S_ + j];
  size_t so = ((size_t)bh * S_ + src) * 64 + d;
  size_t dd = ((size_t)bh * keep + j) * 64 + d;
  KG[dd] = KH[so];
  VG[dd] = VH[so];
}

// ---------------------------------------------------------------------------
// Flash attention over gathered keys; bf16 out. K-chunks async when full.
// ---------------------------------------------------------------------------
__global__ __launch_bounds__(256) void flash_attn_kernel(
    const bf16_t* __restrict__ QH, const bf16_t* __restrict__ KG,
    const bf16_t* __restrict__ VG, bf16_t* __restrict__ O,
    int S_, int keep, float scale) {
  __shared__ alignas(16) bf16_t Qs[128 * 64];   // [q][d]
  __shared__ alignas(16) bf16_t Kc[32 * 64];    // [key][d]
  __shared__ alignas(16) bf16_t Vc[64 * 34];    // [d][key] (padded)
  __shared__ alignas(16) bf16_t Pw[8][16 * 32]; // per-wave P tile [row][key]

  int nq = S_ / 128;
  int qt = blockIdx.x % nq, bh = blockIdx.x / nq;
  int b = bh >> 4, h = bh & 15;
  const bf16_t* Q  = QH + ((size_t)bh * S_ + qt * 128) * 64;
  const bf16_t* Kp = KG + (size_t)bh * keep * 64;
  const bf16_t* Vp = VG + (size_t)bh * keep * 64;
  int tid = threadIdx.x, lane = tid & 31, wave = tid >> 5;
  const int lcol = lane & 15, rh = (lane >> 4) * 8;

  async_copy_2d(Q, 64, Qs, 64, 128, 64, tid);

  v8f o0 = v8f_zero(), o1 = v8f_zero(), o2 = v8f_zero(), o3 = v8f_zero();
  float m_r[8], l_r[8];
#pragma unroll
  for (int r = 0; r < 8; ++r) { m_r[r] = -1e30f; l_r[r] = 0.0f; }

  for (int kc = 0; kc < keep; kc += 32) {
    __syncthreads();
    if (kc + 32 <= keep) {  // uniform: async K chunk, manual V transpose-scatter
      async_copy_2d(Kp + (size_t)kc * 64, 64, Kc, 64, 32, 64, tid);
      for (int i = tid; i < 32 * 64; i += 256) {
        int kk = i >> 6, d = i & 63;
        Vc[d * 34 + kk] = Vp[(size_t)(kc + kk) * 64 + d];
      }
    } else {                // tail: guarded zero-fill copy
      for (int i = tid; i < 32 * 64; i += 256) {
        int kk = i >> 6, d = i & 63;
        bool ok = (kc + kk) < keep;
        Kc[kk * 64 + d] = ok ? Kp[(size_t)(kc + kk) * 64 + d] : (bf16_t)0.0f;
        Vc[d * 34 + kk] = ok ? Vp[(size_t)(kc + kk) * 64 + d] : (bf16_t)0.0f;
      }
    }
    wait_async_all();
    __syncthreads();

    v16bf a0 = load_a_frag(Qs, 64, wave * 16, 0, lane);
    v16bf a1 = load_a_frag(Qs, 64, wave * 16, 32, lane);
    v8f s0 = v8f_zero(), s1 = v8f_zero();
    {
      v16bf b0 = load_b_frag(Kc, 64, 0, 0, lane);
      v16bf b1 = load_b_frag(Kc, 64, 0, 32, lane);
      s0 = WMMA_BF16(a0, b0, s0);
      s0 = WMMA_BF16(a1, b1, s0);
      b0 = load_b_frag(Kc, 64, 16, 0, lane);
      b1 = load_b_frag(Kc, 64, 16, 32, lane);
      s1 = WMMA_BF16(a0, b0, s1);
      s1 = WMMA_BF16(a1, b1, s1);
    }
    bool ok0 = (kc + lcol) < keep;
    bool ok1 = (kc + 16 + lcol) < keep;
#pragma unroll
    for (int r = 0; r < 8; ++r) {
      s0[r] = ok0 ? s0[r] * scale : -1e30f;
      s1[r] = ok1 ? s1[r] * scale : -1e30f;
    }
    // online softmax: a row's 16 values live in one 16-lane half per VGPR idx
#pragma unroll
    for (int r = 0; r < 8; ++r) {
      float mxv = fmaxf(s0[r], s1[r]);
#pragma unroll
      for (int off = 1; off < 16; off <<= 1) mxv = fmaxf(mxv, __shfl_xor(mxv, off, 32));
      float mnew  = fmaxf(m_r[r], mxv);
      float alpha = __expf(m_r[r] - mnew);
      float p0 = __expf(s0[r] - mnew);
      float p1 = __expf(s1[r] - mnew);
      float rs = p0 + p1;
#pragma unroll
      for (int off = 1; off < 16; off <<= 1) rs += __shfl_xor(rs, off, 32);
      l_r[r] = alpha * l_r[r] + rs;
      m_r[r] = mnew;
      o0[r] *= alpha; o1[r] *= alpha; o2[r] *= alpha; o3[r] *= alpha;
      int rl = r + rh;
      Pw[wave][rl * 32 + lcol]      = (bf16_t)p0;
      Pw[wave][rl * 32 + 16 + lcol] = (bf16_t)p1;
    }
    // O += P @ V  (per-wave LDS round-trip re-lays C layout into A layout)
    v16bf ap = load_a_frag(&Pw[wave][0], 32, 0, 0, lane);
    o0 = WMMA_BF16(ap, load_b_frag(Vc, 34, 0,  0, lane), o0);
    o1 = WMMA_BF16(ap, load_b_frag(Vc, 34, 16, 0, lane), o1);
    o2 = WMMA_BF16(ap, load_b_frag(Vc, 34, 32, 0, lane), o2);
    o3 = WMMA_BF16(ap, load_b_frag(Vc, 34, 48, 0, lane), o3);
  }

#pragma unroll
  for (int r = 0; r < 8; ++r) {
    float inv = 1.0f / l_r[r];
    int row = qt * 128 + wave * 16 + r + rh;
    size_t base = ((size_t)b * S_ + row) * 1024 + (size_t)h * 64;
    O[base + 0  + lcol] = (bf16_t)(o0[r] * inv);
    O[base + 16 + lcol] = (bf16_t)(o1[r] * inv);
    O[base + 32 + lcol] = (bf16_t)(o2[r] * inv);
    O[base + 48 + lcol] = (bf16_t)(o3[r] * inv);
  }
}

// ---------------------------------------------------------------------------
// Host orchestration
// ---------------------------------------------------------------------------
extern "C" void kernel_launch(void* const* d_in, const int* in_sizes, int n_in,
                              void* d_out, int out_size, void* d_ws, size_t ws_size,
                              hipStream_t stream) {
  (void)in_sizes; (void)n_in; (void)out_size; (void)ws_size;
  const int B = 2, S = 2048, EMB = 1024, HID = 1024, NH = 16, NL = 4;
  const int M = B * S, H4 = 4 * HID;

  const int*   ids  = (const int*)d_in[0];
  const float* tok  = (const float*)d_in[1];
  const float* pos  = (const float*)d_in[2];
  const float* ipw  = (const float*)d_in[3];
  const float* ipb  = (const float*)d_in[4];
  const float* Wq   = (const float*)d_in[5];
  const float* bq   = (const float*)d_in[6];
  const float* Wk   = (const float*)d_in[7];
  const float* bk   = (const float*)d_in[8];
  const float* Wv   = (const float*)d_in[9];
  const float* bv   = (const float*)d_in[10];
  const float* Wo   = (const float*)d_in[11];
  const float* bo   = (const float*)d_in[12];
  const float* W1   = (const float*)d_in[13];
  const float* b1   = (const float*)d_in[14];
  const float* W2   = (const float*)d_in[15];
  const float* b2   = (const float*)d_in[16];
  const float* ln1s = (const float*)d_in[17];
  const float* ln1b = (const float*)d_in[18];
  const float* ln2s = (const float*)d_in[19];
  const float* ln2b = (const float*)d_in[20];

  char* w = (char*)d_ws;
  auto alloc = [&](size_t bytes) -> char* {
    char* p = w; w += (bytes + 255) & ~(size_t)255; return p;
  };
  // bf16 transposed weights
  bf16_t* IPWt = (bf16_t*)alloc((size_t)EMB * HID * 2);
  bf16_t* WQt  = (bf16_t*)alloc((size_t)NL * HID * HID * 2);
  bf16_t* WKt  = (bf16_t*)alloc((size_t)NL * HID * HID * 2);
  bf16_t* WVt  = (bf16_t*)alloc((size_t)NL * HID * HID * 2);
  bf16_t* WOt  = (bf16_t*)alloc((size_t)NL * HID * HID * 2);
  bf16_t* W1t  = (bf16_t*)alloc((size_t)NL * HID * H4 * 2);
  bf16_t* W2t  = (bf16_t*)alloc((size_t)NL * H4 * HID * 2);
  // activations
  float*  X    = (float*)alloc((size_t)M * HID * 4);   // residual stream (f32)
  bf16_t* XEb  = (bf16_t*)alloc((size_t)M * HID * 2);  // embed out
  bf16_t* Hb   = (bf16_t*)alloc((size_t)M * HID * 2);  // LN out
  bf16_t* AOb  = (bf16_t*)alloc((size_t)M * HID * 2);  // attn out
  bf16_t* MIDb = (bf16_t*)alloc((size_t)M * H4 * 2);   // gelu out
  bf16_t* QH   = (bf16_t*)alloc((size_t)M * HID * 2);
  bf16_t* KH   = (bf16_t*)alloc((size_t)M * HID * 2);
  bf16_t* VH   = (bf16_t*)alloc((size_t)M * HID * 2);
  bf16_t* KG   = (bf16_t*)alloc((size_t)M * HID * 2);
  bf16_t* VG   = (bf16_t*)alloc((size_t)M * HID * 2);
  float*  IMP  = (float*)alloc((size_t)B * S * 4);
  int*    IDX  = (int*)alloc((size_t)B * S * 4);

  dim3 blk(256);
  auto cvt = [&](const float* src, bf16_t* dst, int K, int N) {
    transpose_w_kernel<<<dim3((K * N + 255) / 256), blk, 0, stream>>>(src, dst, K, N);
  };

  // One-time (per launch) weight convert+transpose to bf16 [n][k]
  cvt(ipw, IPWt, EMB, HID);
  for (int i = 0; i < NL; ++i) {
    cvt(Wq + (size_t)i * HID * HID, WQt + (size_t)i * HID * HID, HID, HID);
    cvt(Wk + (size_t)i * HID * HID, WKt + (size_t)i * HID * HID, HID, HID);
    cvt(Wv + (size_t)i * HID * HID, WVt + (size_t)i * HID * HID, HID, HID);
    cvt(Wo + (size_t)i * HID * HID, WOt + (size_t)i * HID * HID, HID, HID);
    cvt(W1 + (size_t)i * HID * H4,  W1t + (size_t)i * HID * H4,  HID, H4);
    cvt(W2 + (size_t)i * H4 * HID,  W2t + (size_t)i * H4 * HID,  H4,  HID);
  }

  embed_kernel<<<dim3((M * EMB + 255) / 256), blk, 0, stream>>>(ids, tok, pos, XEb, S, EMB);
  gemm_async_kernel<0><<<dim3(HID / 128, M / 128), blk, 0, stream>>>(
      XEb, IPWt, ipb, nullptr, X, nullptr, M, HID, EMB, S);

  for (int i = 0; i < NL; ++i) {
    size_t wof = (size_t)i * HID * HID;
    size_t bof = (size_t)i * HID;

    layernorm_kernel<<<dim3(M), blk, 0, stream>>>(X, ln1s + bof, ln1b + bof, Hb, HID);
    gemm_async_kernel<3><<<dim3(HID / 128, M / 128), blk, 0, stream>>>(
        Hb, WQt + wof, bq + bof, nullptr, nullptr, QH, M, HID, HID, S);
    gemm_async_kernel<3><<<dim3(HID / 128, M / 128), blk, 0, stream>>>(
        Hb, WKt + wof, bk + bof, nullptr, nullptr, KH, M, HID, HID, S);
    gemm_async_kernel<3><<<dim3(HID / 128, M / 128), blk, 0, stream>>>(
        Hb, WVt + wof, bv + bof, nullptr, nullptr, VH, M, HID, HID, S);

    double ratio = 1.0 - (double)i * 0.8 / (double)(NL - 1);
    int keep = (int)(ratio * (double)S);
    if (keep < 1) keep = 1;
    if (keep > S) keep = S;

    const bf16_t* KA = KH;
    const bf16_t* VA = VH;
    if (keep < S) {
      hipMemsetAsync(IMP, 0, (size_t)B * S * 4, stream);
      importance_kernel<<<dim3(B * NH * (S / 128)), blk, 0, stream>>>(QH, KH, IMP, S);
      topk_kernel<<<dim3(B), blk, 0, stream>>>(IMP, IDX, S, keep);
      gather_kv_kernel<<<dim3((B * NH * keep * 64 + 255) / 256), blk, 0, stream>>>(
          KH, VH, IDX, KG, VG, S, keep);
      KA = KG; VA = VG;
    }
    flash_attn_kernel<<<dim3(B * NH * (S / 128)), blk, 0, stream>>>(
        QH, KA, VA, AOb, S, keep, 0.125f);
    gemm_async_kernel<1><<<dim3(HID / 128, M / 128), blk, 0, stream>>>(
        AOb, WOt + wof, bo + bof, X, X, nullptr, M, HID, HID, S);

    layernorm_kernel<<<dim3(M), blk, 0, stream>>>(X, ln2s + bof, ln2b + bof, Hb, HID);
    gemm_async_kernel<2><<<dim3(H4 / 128, M / 128), blk, 0, stream>>>(
        Hb, W1t + (size_t)i * HID * H4, b1 + (size_t)i * H4, nullptr, nullptr, MIDb,
        M, H4, HID, S);
    gemm_async_kernel<1><<<dim3(HID / 128, M / 128), blk, 0, stream>>>(
        MIDb, W2t + (size_t)i * H4 * HID, b2 + bof, X, X, nullptr, M, HID, H4, S);
  }

  hipMemcpyAsync(d_out, X, (size_t)M * HID * 4, hipMemcpyDeviceToDevice, stream);
}
